// SokobanNCA_60833916781275
// MI455X (gfx1250) — matrix operand order repaired and checked by reading, not compiled, not run on hardware
//
#include <hip/hip_runtime.h>
#include <cstdint>

// ---------------- problem constants ----------------
#define HC   128     // hidden channels
#define NPIX 100     // 10x10
#define NPAD 112     // 7 N-tiles of 16
#define NT   7
#define PADW 12      // 12x12 zero-halo spatial layout
#define PADP 144
#define BTOT 2048

typedef __bf16 bf16_t;
typedef __attribute__((ext_vector_type(8)))  __bf16 v8bf;
typedef __attribute__((ext_vector_type(16))) __bf16 v16bf;
typedef __attribute__((ext_vector_type(8)))  float  v8f;

// ---------------- d_ws weight layout (bf16 elements) ----------------
#define WP1_OFF 0                         // [128][64]  im2col K = tap*5+c, pad to 64
#define WP2_OFF (WP1_OFF + 128*64)        // [9][128][128] per-tap 1x1 weights
#define WP3_OFF (WP2_OFF + 9*128*128)     // [128][128]
#define WU1_OFF (WP3_OFF + 128*128)
#define WU2_OFF (WU1_OFF + 128*128)
#define WU3_OFF (WU2_OFF + 128*128)
#define WU4_OFF (WU3_OFF + 128*128)       // [16][128], rows 5..15 zero

// ---------------- LDS layout (bytes) ----------------
//  act1  : [144][128] bf16  (halo)         36864
//  act2  : [112][128] bf16                 28672
//  act3  : [112][128] bf16                 28672
//  xim   : [112][64]  bf16 (im2col)        14336
//  xpad  : [144][8]   bf16 (state, halo)    2304
//  xs    : [500]      f32  (state, c*100+p) 2000
//  dxb   : [112][8]   f32  (head output)    3584
#define LDS_BYTES (36864 + 28672 + 28672 + 14336 + 2304 + 2000 + 3584)

// =====================================================================
// Weight prepack: f32 -> bf16, GEMM-friendly layouts in d_ws
// =====================================================================
__global__ void nca_prepack(const float* __restrict__ wp1, const float* __restrict__ wp2,
                            const float* __restrict__ wp3, const float* __restrict__ wu1,
                            const float* __restrict__ wu2, const float* __restrict__ wu3,
                            const float* __restrict__ wu4, bf16_t* __restrict__ ws) {
  int i      = blockIdx.x * blockDim.x + threadIdx.x;
  int stride = gridDim.x * blockDim.x;
  // conv1 im2col weights [128][64]
  for (int idx = i; idx < 128 * 64; idx += stride) {
    int m = idx >> 6, k = idx & 63;
    float v = 0.0f;
    if (k < 45) {
      int t = k / 5, c = k - t * 5, ky = t / 3, kx = t - ky * 3;
      v = wp1[((m * 5 + c) * 3 + ky) * 3 + kx];
    }
    ws[WP1_OFF + idx] = (bf16_t)v;
  }
  // conv2 per-tap weights [9][128][128]
  for (int idx = i; idx < 9 * 128 * 128; idx += stride) {
    int t = idx >> 14, mk = idx & 16383, m = mk >> 7, k = mk & 127;
    int ky = t / 3, kx = t - ky * 3;
    ws[WP2_OFF + idx] = (bf16_t)wp2[((m * 128 + k) * 3 + ky) * 3 + kx];
  }
  // 1x1 convs [128][128]
  for (int idx = i; idx < 128 * 128; idx += stride) {
    ws[WP3_OFF + idx] = (bf16_t)wp3[idx];
    ws[WU1_OFF + idx] = (bf16_t)wu1[idx];
    ws[WU2_OFF + idx] = (bf16_t)wu2[idx];
    ws[WU3_OFF + idx] = (bf16_t)wu3[idx];
  }
  // head [16][128], rows >=5 zero
  for (int idx = i; idx < 16 * 128; idx += stride) {
    int m = idx >> 7, k = idx & 127;
    ws[WU4_OFF + idx] = (bf16_t)(m < 5 ? wu4[m * 128 + k] : 0.0f);
  }
}

// =====================================================================
// WMMA fragment helpers (CDNA5 16x16x32 bf16 layouts)
// =====================================================================
// A (16x32, M row-major, K contiguous): lanes 0-15 hold K 0-7 & 16-23,
// lanes 16-31 hold K 8-15 & 24-31, row M = lane&15 (caller adds row base).
__device__ __forceinline__ v16bf load_afrag(const bf16_t* rowbase, int kt, int lane) {
  int koff = kt * 32 + ((lane & 16) ? 8 : 0);
  union { v16bf v; v8bf h[2]; } u;
  u.h[0] = *(const v8bf*)(rowbase + koff);
  u.h[1] = *(const v8bf*)(rowbase + koff + 16);
  return u.v;
}
// B (32x16, column-major per pixel, K contiguous): lanes 0-15 hold K 0-7 & 16-23,
// lanes 16-31 hold K 8-15 & 24-31, column N = lane&15 (caller passes column base).
__device__ __forceinline__ v16bf load_bfrag(const bf16_t* colbase, int kt, int lane) {
  int koff = kt * 32 + ((lane & 16) ? 8 : 0);
  union { v16bf v; v8bf h[2]; } u;
  u.h[0] = *(const v8bf*)(colbase + koff);
  u.h[1] = *(const v8bf*)(colbase + koff + 16);
  return u.v;
}
__device__ __forceinline__ v8f wmma_bf16(v16bf a, v16bf b, v8f c) {
  return __builtin_amdgcn_wmma_f32_16x16x32_bf16(false, a, false, b, (short)0, c,
                                                 false, false);
}

// C/D (16x16 f32): lane<16 -> M=0..7 (vgpr r = M r), lane>=16 -> M=8..15; N=lane&15.
// bias + relu + cvt bf16 + one b128 store per (lane, n-tile).
__device__ __forceinline__ void store_tiles(const v8f* acc, const float* __restrict__ biasG,
                                            bf16_t* dst, int wave, int lane, bool halo) {
  int half = lane >> 4;
  int l15  = lane & 15;
  const float* bp = biasG + wave * 16 + half * 8;
  float bb[8];
#pragma unroll
  for (int r = 0; r < 8; ++r) bb[r] = bp[r];
#pragma unroll
  for (int nt = 0; nt < NT; ++nt) {
    int n = nt * 16 + l15;
    if (n < NPIX) {
      int row = n;
      if (halo) { int y = n / 10, x = n - y * 10; row = (y + 1) * PADW + (x + 1); }
      v8bf o;
#pragma unroll
      for (int r = 0; r < 8; ++r) {
        float v = acc[nt][r] + bb[r];
        v = fmaxf(v, 0.0f);
        o[r] = (bf16_t)v;
      }
      *(v8bf*)(dst + row * HC + wave * 16 + half * 8) = o;
    }
  }
}

// 128->128 1x1 conv as one 128 x 112 x 128 GEMM (+bias, relu)
__device__ __forceinline__ void conv1x1_gemm(const bf16_t* __restrict__ W,
                                             const float* __restrict__ biasG,
                                             const bf16_t* src, bf16_t* dst,
                                             int wave, int lane) {
  int l15 = lane & 15;
  v8f acc[NT];
#pragma unroll
  for (int nt = 0; nt < NT; ++nt) acc[nt] = v8f{0.f, 0.f, 0.f, 0.f, 0.f, 0.f, 0.f, 0.f};
#pragma unroll
  for (int kt = 0; kt < 4; ++kt) {
    v16bf A = load_afrag(W + (wave * 16 + l15) * HC, kt, lane);
#pragma unroll
    for (int nt = 0; nt < NT; ++nt) {
      v16bf Bm = load_bfrag(src + (nt * 16 + l15) * HC, kt, lane);
      acc[nt] = wmma_bf16(A, Bm, acc[nt]);
    }
  }
  store_tiles(acc, biasG, dst, wave, lane, false);
}

// =====================================================================
// Main fused NCA kernel: one workgroup per batch element, all steps
// =====================================================================
__global__ __launch_bounds__(256)
void nca_kernel(const float* __restrict__ x_in,
                const float* __restrict__ b_p1, const float* __restrict__ b_p2,
                const float* __restrict__ b_p3, const float* __restrict__ b_u1,
                const float* __restrict__ b_u2, const float* __restrict__ b_u3,
                const float* __restrict__ b_u4, const int* __restrict__ steps_p,
                const bf16_t* __restrict__ ws, float* __restrict__ out) {
  extern __shared__ char smem[];
  bf16_t* act1 = (bf16_t*)smem;              // [144][128]
  bf16_t* act2 = act1 + PADP * HC;           // [112][128]
  bf16_t* act3 = act2 + NPAD * HC;           // [112][128]
  bf16_t* xim  = act3 + NPAD * HC;           // [112][64]
  bf16_t* xpad = xim + NPAD * 64;            // [144][8]
  float*  xs   = (float*)(xpad + PADP * 8);  // [500] = [c][p]
  float*  dxb  = xs + 500;                   // [112][8]

  const int tid  = threadIdx.x;
  const int lane = tid & 31;
  const int wave = tid >> 5;
  const int l15  = lane & 15;
  const int b    = blockIdx.x;
  const int steps = steps_p[0];

  const bf16_t* WP1 = ws + WP1_OFF;
  const bf16_t* WP2 = ws + WP2_OFF;
  const bf16_t* WP3 = ws + WP3_OFF;
  const bf16_t* WU1 = ws + WU1_OFF;
  const bf16_t* WU2 = ws + WU2_OFF;
  const bf16_t* WU3 = ws + WU3_OFF;
  const bf16_t* WU4 = ws + WU4_OFF;

  // zero-init all LDS once: halos / pad columns / pad rows stay zero forever
  for (int i = tid; i < LDS_BYTES / 4; i += 256) ((uint32_t*)smem)[i] = 0u;
  __syncthreads();

  // load initial state
  for (int i = tid; i < 500; i += 256) xs[i] = x_in[(size_t)b * 500 + i];
  __syncthreads();
  for (int p = tid; p < NPIX; p += 256) {
    int y = p / 10, x = p - y * 10;
    int pi = (y + 1) * PADW + (x + 1);
#pragma unroll
    for (int c = 0; c < 5; ++c) xpad[pi * 8 + c] = (bf16_t)xs[c * 100 + p];
  }
  __syncthreads();

  float* states = out + (size_t)BTOT * 500;  // d_out = x_final ++ states

  for (int step = 0; step < steps; ++step) {
    // ---- 1) im2col for conv1 (K = tap*5 + c, 45 used of 64) ----
    for (int idx = tid; idx < NPIX * 9; idx += 256) {
      int n = idx / 9, t = idx - n * 9;
      int y = n / 10, x = n - y * 10;
      int ky = t / 3, kx = t - ky * 3;
      int pi = (y + ky) * PADW + (x + kx);
#pragma unroll
      for (int c = 0; c < 5; ++c) xim[n * 64 + t * 5 + c] = xpad[pi * 8 + c];
    }
    __syncthreads();

    // ---- 2) conv1: 5->128, K=64 im2col GEMM, relu, store into halo buffer ----
    {
      v8f acc[NT];
#pragma unroll
      for (int nt = 0; nt < NT; ++nt) acc[nt] = v8f{0.f, 0.f, 0.f, 0.f, 0.f, 0.f, 0.f, 0.f};
#pragma unroll
      for (int kt = 0; kt < 2; ++kt) {
        v16bf A = load_afrag(WP1 + (wave * 16 + l15) * 64, kt, lane);
#pragma unroll
        for (int nt = 0; nt < NT; ++nt) {
          v16bf Bm = load_bfrag(xim + (nt * 16 + l15) * 64, kt, lane);
          acc[nt] = wmma_bf16(A, Bm, acc[nt]);
        }
      }
      store_tiles(acc, b_p1, act1, wave, lane, true);
    }
    __syncthreads();

    // ---- 3) conv2: 128->128 3x3 = 9 shifted 1x1 GEMMs over halo buffer ----
    {
      int py[NT], px[NT]; bool valid[NT];
#pragma unroll
      for (int nt = 0; nt < NT; ++nt) {
        int n = nt * 16 + l15;
        valid[nt] = (n < NPIX);
        py[nt] = n / 10; px[nt] = n - py[nt] * 10;
      }
      v8f acc[NT];
#pragma unroll
      for (int nt = 0; nt < NT; ++nt) acc[nt] = v8f{0.f, 0.f, 0.f, 0.f, 0.f, 0.f, 0.f, 0.f};
      for (int t = 0; t < 9; ++t) {           // tap loop kept rolled
        int ky = t / 3, kx = t - ky * 3;
#pragma unroll
        for (int kt = 0; kt < 4; ++kt) {
          v16bf A = load_afrag(WP2 + (size_t)(t * 128 + wave * 16 + l15) * HC, kt, lane);
#pragma unroll
          for (int nt = 0; nt < NT; ++nt) {
            int pi = valid[nt] ? ((py[nt] + ky) * PADW + px[nt] + kx) : 0; // halo -> zeros
            v16bf Bm = load_bfrag(act1 + pi * HC, kt, lane);
            acc[nt] = wmma_bf16(A, Bm, acc[nt]);
          }
        }
      }
      store_tiles(acc, b_p2, act2, wave, lane, false);
    }
    __syncthreads();

    // ---- 4) 1x1 chain, ping-pong act2 <-> act3 ----
    conv1x1_gemm(WP3, b_p3, act2, act3, wave, lane); __syncthreads();
    conv1x1_gemm(WU1, b_u1, act3, act2, wave, lane); __syncthreads();
    conv1x1_gemm(WU2, b_u2, act2, act3, wave, lane); __syncthreads();
    conv1x1_gemm(WU3, b_u3, act3, act2, wave, lane); __syncthreads();

    // ---- 5) head: 128->5 (M padded to 16, rows 5..15 zero). 7 waves, 1 n-tile each ----
    if (wave < 7) {
      v8f acc = v8f{0.f, 0.f, 0.f, 0.f, 0.f, 0.f, 0.f, 0.f};
#pragma unroll
      for (int kt = 0; kt < 4; ++kt) {
        v16bf A  = load_afrag(WU4 + l15 * HC, kt, lane);
        v16bf Bm = load_bfrag(act2 + (wave * 16 + l15) * HC, kt, lane);
        acc = wmma_bf16(A, Bm, acc);
      }
      int n = wave * 16 + l15;
      if (lane < 16 && n < NPIX) {            // lanes<16 hold M=0..7 (ch 0..4 valid)
#pragma unroll
        for (int r = 0; r < 8; ++r) {
          float bb = (r < 5) ? b_u4[r] : 0.0f;
          dxb[n * 8 + r] = acc[r] + bb;
        }
      }
    }
    __syncthreads();

    // ---- 6) residual + tempered softmax over C=5, update state ----
    {
      float temp = 1.0f - (float)step / (float)steps;
      temp = fmaxf(temp, 0.5f);
      float invt = 1.0f / temp;
      for (int p = tid; p < NPIX; p += 256) {
        float v[5]; float m = -3.0e38f;
#pragma unroll
        for (int c = 0; c < 5; ++c) { v[c] = xs[c * 100 + p] + dxb[p * 8 + c]; m = fmaxf(m, v[c]); }
        float e[5]; float s = 0.0f;
#pragma unroll
        for (int c = 0; c < 5; ++c) { e[c] = __expf((v[c] - m) * invt); s += e[c]; }
        float is = 1.0f / s;
        int y = p / 10, x = p - y * 10;
        int pi = (y + 1) * PADW + (x + 1);
#pragma unroll
        for (int c = 0; c < 5; ++c) {
          float o = e[c] * is;
          xs[c * 100 + p]   = o;
          xpad[pi * 8 + c]  = (bf16_t)o;
        }
      }
    }
    __syncthreads();

    // ---- 7) emit this step's state (coalesced) ----
    {
      float* sp = states + ((size_t)step * BTOT + b) * 500;
      for (int i = tid; i < 500; i += 256) sp[i] = xs[i];
    }
    __syncthreads();
  }

  // x_final
  for (int i = tid; i < 500; i += 256) out[(size_t)b * 500 + i] = xs[i];
}

// =====================================================================
extern "C" void kernel_launch(void* const* d_in, const int* in_sizes, int n_in,
                              void* d_out, int out_size, void* d_ws, size_t ws_size,
                              hipStream_t stream) {
  const float* x   = (const float*)d_in[0];
  const float* wp1 = (const float*)d_in[1];
  const float* bp1 = (const float*)d_in[2];
  const float* wp2 = (const float*)d_in[3];
  const float* bp2 = (const float*)d_in[4];
  const float* wp3 = (const float*)d_in[5];
  const float* bp3 = (const float*)d_in[6];
  const float* wu1 = (const float*)d_in[7];
  const float* bu1 = (const float*)d_in[8];
  const float* wu2 = (const float*)d_in[9];
  const float* bu2 = (const float*)d_in[10];
  const float* wu3 = (const float*)d_in[11];
  const float* bu3 = (const float*)d_in[12];
  const float* wu4 = (const float*)d_in[13];
  const float* bu4 = (const float*)d_in[14];
  const int* steps = (const int*)d_in[15];
  bf16_t* ws = (bf16_t*)d_ws;

  nca_prepack<<<128, 256, 0, stream>>>(wp1, wp2, wp3, wu1, wu2, wu3, wu4, ws);

  (void)hipFuncSetAttribute((const void*)nca_kernel,
                            hipFuncAttributeMaxDynamicSharedMemorySize, LDS_BYTES);
  nca_kernel<<<BTOT, 256, LDS_BYTES, stream>>>(x, bp1, bp2, bp3, bu1, bu2, bu3, bu4,
                                               steps, ws, (float*)d_out);
}
